// FPM_69475390980590
// MI455X (gfx1250) — matrix operand order repaired and checked by reading, not compiled, not run on hardware
//
#include <hip/hip_runtime.h>
#include <hip/hip_bf16.h>

// Problem constants (B=4, C=256, H=W=64)
#define BATCH 4
#define CC    256
#define HWN   4096
#define LOG2E 1.44269504088896f

typedef __attribute__((ext_vector_type(16))) __bf16 v16bf;
typedef __attribute__((ext_vector_type(8)))  __bf16 v8bf;
typedef __attribute__((ext_vector_type(8)))  float  v8f;

// Hardware exp2: v_exp_f32 is natively base-2.
static __device__ __forceinline__ float fast_exp2(float x) {
    return __builtin_amdgcn_exp2f(x);
}
static __device__ __forceinline__ unsigned short bfbits(float f) {
    __bf16 h = (__bf16)f;
    return __builtin_bit_cast(unsigned short, h);
}

// Build a v16bf WMMA operand from two 16-byte chunks.
static __device__ __forceinline__ v16bf ld_op(const __bf16* lo, const __bf16* hi) {
    v8bf a = *(const v8bf*)lo;
    v8bf b = *(const v8bf*)hi;
    return __builtin_shufflevector(a, b, 0,1,2,3,4,5,6,7,8,9,10,11,12,13,14,15);
}
// A operand (lane = row, K chunk k): base = row*CC + lhi*16, +k*32, 16 contiguous
static __device__ __forceinline__ v16bf ldA(const __bf16* p) { return ld_op(p, p + 8); }
// B operand (lane = col, K chunk k): base = col_row + k*32 + lhi*16, 16 contiguous
static __device__ __forceinline__ v16bf ldB(const __bf16* p) { return ld_op(p, p + 8); }

#define WMMA_BF16(a, b, c) \
    __builtin_amdgcn_wmma_f32_16x16x32_bf16(false, (a), false, (b), (short)0, (c), false, false)

// ---------------------------------------------------------------------------
// Kernel 0: fuse = f_m + f_d -> bf16 transposed fT[b][n][c];  fd -> bf16 [b][c][n]
// 64n x 32c tile; all global traffic is b64 loads / b32 stores.
// ---------------------------------------------------------------------------
__global__ void __launch_bounds__(256)
k_prep(const float* __restrict__ fm, const float* __restrict__ fd,
       unsigned short* __restrict__ fT_, unsigned short* __restrict__ fdb_) {
    __shared__ __bf16 tl[64][34];                 // [n_local][c_local], padded

    const int bx = blockIdx.x;                    // b*8 + ctile
    const int b  = bx >> 3;
    const int c0 = (bx & 7) * 32;
    const int n0 = blockIdx.y * 64;
    const int tx = threadIdx.x & 31;              // n-pair index
    const int ty = threadIdx.x >> 5;              // 0..7

    const size_t base = (size_t)b * CC * HWN;
#pragma unroll
    for (int i = 0; i < 4; ++i) {
        const int cl = ty + i * 8;                // 0..31
        const size_t idx = base + (size_t)(c0 + cl) * HWN + n0 + tx * 2;
        const float2 m2 = *(const float2*)(fm + idx);
        const float2 d2 = *(const float2*)(fd + idx);
        *(unsigned int*)(fdb_ + idx) =
            (unsigned int)bfbits(d2.x) | ((unsigned int)bfbits(d2.y) << 16);
        tl[tx * 2 + 0][cl] = (__bf16)(m2.x + d2.x);
        tl[tx * 2 + 1][cl] = (__bf16)(m2.y + d2.y);
    }
    __syncthreads();
    const size_t baseT = (size_t)b * HWN * CC;
    const int cp = threadIdx.x & 15;              // c-pair index
    const int rw = threadIdx.x >> 4;              // 0..15
#pragma unroll
    for (int j = 0; j < 4; ++j) {
        const int nl = rw + j * 16;               // 0..63
        const unsigned int v = *(const unsigned int*)&tl[nl][cp * 2];
        *(unsigned int*)(fT_ + baseT + (size_t)(n0 + nl) * CC + c0 + cp * 2) = v;
    }
}

// ---------------------------------------------------------------------------
// Kernel 1: softmax stats (base-2). S[n,m] = sum_c fT[n][c]*fT[m][c].
// Workgroup = 128 n-rows (8 waves x 16). m streamed in 64-blocks staged in
// LDS; B fetches software-pipelined one K-step ahead of the WMMA group.
// ---------------------------------------------------------------------------
__global__ void __launch_bounds__(256)
k_stats(const unsigned short* __restrict__ fT_,
        float* __restrict__ Mv, float* __restrict__ Zv) {
    __shared__ __align__(16) __bf16 Bls[64][264];   // [m][c], padded stride

    const __bf16* fT = (const __bf16*)fT_;
    const int b    = blockIdx.x;
    const int nblk = blockIdx.y;
    const int wave = threadIdx.x >> 5;
    const int lane = threadIdx.x & 31;
    const int lmod = lane & 15;
    const int lhi  = lane >> 4;                   // 0: rows 0-7, 1: rows 8-15
    const int n0   = nblk * 128 + wave * 16;

    const __bf16* base = fT + (size_t)b * HWN * CC;

    // Preload A operand (16 n-rows x 256 K=c): 8 x v16bf = 64 VGPRs.
    const __bf16* arow = base + (size_t)(n0 + lmod) * CC + lhi * 16;
    v16bf A[8];
#pragma unroll
    for (int k = 0; k < 8; ++k) A[k] = ldA(arow + k * 32);

    float runM[8], runZ[8];
#pragma unroll
    for (int r = 0; r < 8; ++r) { runM[r] = -3.0e38f; runZ[r] = 0.0f; }

    const int ldrow = threadIdx.x >> 2;
    const int ldcol = (threadIdx.x & 3) * 64;

    for (int mb = 0; mb < HWN / 64; ++mb) {
        __syncthreads();                          // previous block's reads done
        {   // cooperative stage of 64 m-rows x 256 c into LDS
            const __bf16* src = base + (size_t)(mb * 64 + ldrow) * CC + ldcol;
            __bf16* dst = &Bls[ldrow][ldcol];
#pragma unroll
            for (int j = 0; j < 8; ++j)
                *(v8bf*)(dst + j * 8) = *(const v8bf*)(src + j * 8);
            if (mb + 1 < HWN / 64)
                __builtin_prefetch(base + (size_t)((mb + 1) * 64 + ldrow) * CC + ldcol, 0, 0);
        }
        __syncthreads();

        // 4 independent 16x16 tiles; B fetch pipelined one K-step ahead.
        v8f s0 = {}, s1 = {}, s2 = {}, s3 = {};
        v16bf b0 = ldB(&Bls[ 0 + lmod][lhi * 16]);
        v16bf b1 = ldB(&Bls[16 + lmod][lhi * 16]);
        v16bf b2 = ldB(&Bls[32 + lmod][lhi * 16]);
        v16bf b3 = ldB(&Bls[48 + lmod][lhi * 16]);
#pragma unroll
        for (int k = 0; k < 8; ++k) {
            v16bf t0, t1, t2, t3;
            if (k < 7) {
                const int co = (k + 1) * 32 + lhi * 16;
                t0 = ldB(&Bls[ 0 + lmod][co]);
                t1 = ldB(&Bls[16 + lmod][co]);
                t2 = ldB(&Bls[32 + lmod][co]);
                t3 = ldB(&Bls[48 + lmod][co]);
            }
            s0 = WMMA_BF16(A[k], b0, s0);
            s1 = WMMA_BF16(A[k], b1, s1);
            s2 = WMMA_BF16(A[k], b2, s2);
            s3 = WMMA_BF16(A[k], b3, s3);
            if (k < 7) { b0 = t0; b1 = t1; b2 = t2; b3 = t3; }
        }
        // Per-lane online softmax update (base 2, no cross-lane ops here).
        v8f st[4] = { s0, s1, s2, s3 };
#pragma unroll
        for (int mt = 0; mt < 4; ++mt) {
#pragma unroll
            for (int r = 0; r < 8; ++r) {
                const float v  = st[mt][r] * LOG2E;
                const float nm = fmaxf(runM[r], v);
                runZ[r] = runZ[r] * fast_exp2(runM[r] - nm) + fast_exp2(v - nm);
                runM[r] = nm;
            }
        }
    }

    // Merge the 16 per-lane partials of each half-wave (rows shared per half).
#pragma unroll
    for (int r = 0; r < 8; ++r) {
#pragma unroll
        for (int off = 1; off <= 8; off <<= 1) {
            const float oM = __shfl_xor(runM[r], off);
            const float oZ = __shfl_xor(runZ[r], off);
            const float nm = fmaxf(runM[r], oM);
            runZ[r] = runZ[r] * fast_exp2(runM[r] - nm) + oZ * fast_exp2(oM - nm);
            runM[r] = nm;
        }
    }
    if (lmod == 0) {
        const size_t o = (size_t)b * HWN + n0 + lhi * 8;
#pragma unroll
        for (int r = 0; r < 8; ++r) { Mv[o + r] = runM[r]; Zv[o + r] = runZ[r]; }
    }
}

// ---------------------------------------------------------------------------
// Kernel 2: out[c,m] = f_d[c,m] + sum_n fd[c,n] * 2^(s2[n,m]-M2[n]) / Z[n].
// Workgroup owns 64 m-columns and all 256 c-rows; operand fetches pipelined.
// ---------------------------------------------------------------------------
__global__ void __launch_bounds__(256)
k_out(const unsigned short* __restrict__ fT_,
      const unsigned short* __restrict__ fdb_,
      const float* __restrict__ Mv, const float* __restrict__ Zv,
      const float* __restrict__ fd, float* __restrict__ out) {
    __shared__ __align__(16) __bf16 Bls[64][264];   // fixed GEMM1 B: [m][c]
    __shared__ __align__(16) __bf16 Pt[64][136];    // staged P^T: [m][n]
    __shared__ float sM[128], sZ[128];

    const __bf16* fT  = (const __bf16*)fT_;
    const __bf16* fdb = (const __bf16*)fdb_;

    const int b    = blockIdx.x;
    const int m0   = blockIdx.y * 64;
    const int wave = threadIdx.x >> 5;
    const int lane = threadIdx.x & 31;
    const int lmod = lane & 15;
    const int lhi  = lane >> 4;

    const __bf16* ftb  = fT  + (size_t)b * HWN * CC;
    const __bf16* fdbb = fdb + (size_t)b * CC * HWN;

    {   // Stage the workgroup's 64 m-rows of fT (full C) into LDS once.
        const int ldrow = threadIdx.x >> 2;
        const int ldcol = (threadIdx.x & 3) * 64;
        const __bf16* src = ftb + (size_t)(m0 + ldrow) * CC + ldcol;
        __bf16* dst = &Bls[ldrow][ldcol];
#pragma unroll
        for (int j = 0; j < 8; ++j)
            *(v8bf*)(dst + j * 8) = *(const v8bf*)(src + j * 8);
    }
    __syncthreads();

    v8f acc[2][4] = {};   // [c-subtile 0/1][m-subtile 0..3]

    for (int nb = 0; nb < HWN; nb += 128) {
        if (threadIdx.x < 128) {
            sM[threadIdx.x] = Mv[(size_t)b * HWN + nb + threadIdx.x];
            sZ[threadIdx.x] = Zv[(size_t)b * HWN + nb + threadIdx.x];
        }
        // Phase A: recompute S tiles [16n x 64m]; A (global) + B (LDS)
        // fetches pipelined one K-step ahead.
        const __bf16* arow = ftb + (size_t)(nb + wave * 16 + lmod) * CC + lhi * 16;
        v8f s0 = {}, s1 = {}, s2 = {}, s3 = {};
        v16bf av = ldA(arow);
        v16bf b0 = ldB(&Bls[ 0 + lmod][lhi * 16]);
        v16bf b1 = ldB(&Bls[16 + lmod][lhi * 16]);
        v16bf b2 = ldB(&Bls[32 + lmod][lhi * 16]);
        v16bf b3 = ldB(&Bls[48 + lmod][lhi * 16]);
#pragma unroll
        for (int k = 0; k < 8; ++k) {
            v16bf an, t0, t1, t2, t3;
            if (k < 7) {
                an = ldA(arow + (k + 1) * 32);
                const int co = (k + 1) * 32 + lhi * 16;
                t0 = ldB(&Bls[ 0 + lmod][co]);
                t1 = ldB(&Bls[16 + lmod][co]);
                t2 = ldB(&Bls[32 + lmod][co]);
                t3 = ldB(&Bls[48 + lmod][co]);
            }
            s0 = WMMA_BF16(av, b0, s0);
            s1 = WMMA_BF16(av, b1, s1);
            s2 = WMMA_BF16(av, b2, s2);
            s3 = WMMA_BF16(av, b3, s3);
            if (k < 7) { av = an; b0 = t0; b1 = t1; b2 = t2; b3 = t3; }
        }
        __syncthreads();   // sM/sZ visible; previous iteration's Pt reads done

        // Normalize and stage transposed bf16 P in LDS.
        v8f st[4] = { s0, s1, s2, s3 };
#pragma unroll
        for (int mt = 0; mt < 4; ++mt) {
            v8bf pv;
#pragma unroll
            for (int r = 0; r < 8; ++r) {
                const int nl = wave * 16 + lhi * 8 + r;
                pv[r] = (__bf16)(fast_exp2(st[mt][r] * LOG2E - sM[nl]) * (1.0f / sZ[nl]));
            }
            *(v8bf*)&Pt[mt * 16 + lmod][wave * 16 + lhi * 8] = pv;
        }
        __syncthreads();

        // Phase B: acc += fd[c, n-block] x P[n-block, m]; global A pipelined.
        const __bf16* abase = fdbb + (size_t)(wave * 32 + lmod) * HWN + nb + lhi * 16;
        v16bf av0 = ldA(abase);
        v16bf av1 = ldA(abase + (size_t)16 * HWN);
#pragma unroll
        for (int nk = 0; nk < 4; ++nk) {
            v16bf n0v, n1v;
            if (nk < 3) {
                n0v = ldA(abase + (nk + 1) * 32);
                n1v = ldA(abase + (nk + 1) * 32 + (size_t)16 * HWN);
            }
            const int no = nk * 32 + lhi * 16;
            v16bf p0 = ldB(&Pt[ 0 + lmod][no]);
            v16bf p1 = ldB(&Pt[16 + lmod][no]);
            v16bf p2 = ldB(&Pt[32 + lmod][no]);
            v16bf p3 = ldB(&Pt[48 + lmod][no]);
            acc[0][0] = WMMA_BF16(av0, p0, acc[0][0]);
            acc[1][0] = WMMA_BF16(av1, p0, acc[1][0]);
            acc[0][1] = WMMA_BF16(av0, p1, acc[0][1]);
            acc[1][1] = WMMA_BF16(av1, p1, acc[1][1]);
            acc[0][2] = WMMA_BF16(av0, p2, acc[0][2]);
            acc[1][2] = WMMA_BF16(av1, p2, acc[1][2]);
            acc[0][3] = WMMA_BF16(av0, p3, acc[0][3]);
            acc[1][3] = WMMA_BF16(av1, p3, acc[1][3]);
            if (nk < 3) { av0 = n0v; av1 = n1v; }
        }
    }

    // Epilogue: residual add + coalesced store.
#pragma unroll
    for (int t = 0; t < 2; ++t) {
#pragma unroll
        for (int mt = 0; mt < 4; ++mt) {
#pragma unroll
            for (int r = 0; r < 8; ++r) {
                const int c = wave * 32 + t * 16 + lhi * 8 + r;
                const size_t idx = ((size_t)b * CC + c) * HWN + m0 + mt * 16 + lmod;
                out[idx] = acc[t][mt][r] + fd[idx];
            }
        }
    }
}

// ---------------------------------------------------------------------------
extern "C" void kernel_launch(void* const* d_in, const int* in_sizes, int n_in,
                              void* d_out, int out_size, void* d_ws, size_t ws_size,
                              hipStream_t stream) {
    (void)in_sizes; (void)n_in; (void)out_size; (void)ws_size;
    const float* fm = (const float*)d_in[0];
    const float* fd = (const float*)d_in[1];
    float* out = (float*)d_out;

    char* ws = (char*)d_ws;
    unsigned short* fT  = (unsigned short*)ws;                       // 8 MB  [b][n][c] bf16
    unsigned short* fdb = (unsigned short*)(ws + (size_t)8388608);   // 8 MB  [b][c][n] bf16
    float* Mv = (float*)(ws + (size_t)16777216);                     // 64 KB [b][n]
    float* Zv = (float*)(ws + (size_t)16777216 + 65536);             // 64 KB [b][n]

    dim3 g0(32, 64);
    k_prep<<<g0, 256, 0, stream>>>(fm, fd, fT, fdb);

    dim3 g1(BATCH, HWN / 128);
    k_stats<<<g1, 256, 0, stream>>>(fT, Mv, Zv);

    dim3 g2(BATCH, HWN / 64);
    k_out<<<g2, 256, 0, stream>>>(fT, fdb, Mv, Zv, fd, out);
}
